// CorrectionPotential_33079838114388
// MI455X (gfx1250) — matrix operand order repaired and checked by reading, not compiled, not run on hardware
//
#include <hip/hip_runtime.h>
#include <hip/hip_bf16.h>

typedef __attribute__((ext_vector_type(16))) _Float16 v16h;
typedef __attribute__((ext_vector_type(8)))  _Float16 v8h;
typedef __attribute__((ext_vector_type(8)))  float    v8f;

#define WAVES_PER_BLOCK 8
#define CUTOFF_F 6.0f
#define GAMMA_F (10.0f / (CUTOFF_F * CUTOFF_F))
#define CSPACE (CUTOFF_F / 31.0f)   // linspace(0, 6, 32) spacing

__device__ __forceinline__ float fast_tanh(float x) {
#if __has_builtin(__builtin_amdgcn_tanhf)
    // CDNA5 hardware transcendental: v_tanh_f32 (TRANS op, co-executes with WMMA)
    return __builtin_amdgcn_tanhf(x);
#else
    float e = __expf(2.0f * x);
#if __has_builtin(__builtin_amdgcn_rcpf)
    return 1.0f - 2.0f * __builtin_amdgcn_rcpf(e + 1.0f);
#else
    return 1.0f - 2.0f / (e + 1.0f);
#endif
#endif
}

__global__ void zero_kernel(float* out) {
    if (threadIdx.x == 0 && blockIdx.x == 0) out[0] = 0.0f;
}

__global__ __launch_bounds__(256, 1)
void pair_mlp_wmma(const float* __restrict__ R,
                   const int*   __restrict__ Z,
                   const float* __restrict__ emb,
                   const float* __restrict__ W1, const float* __restrict__ b1,
                   const float* __restrict__ W2, const float* __restrict__ b2,
                   const float* __restrict__ W3, const float* __restrict__ b3,
                   float* __restrict__ out, int N)
{
    const int lane = threadIdx.x & 31;
    const int wave = threadIdx.x >> 5;
    const int hi   = lane >> 4;      // K-half selector for A / M-half for C rows
    const int n0   = lane & 15;      // column-in-tile (B/C/D) and pair row m (A)

    // ---- Load W1/W2 B-fragments into registers (f32 -> f16 on the fly) ----
    // B layout (16x16x32 f16): lane l holds col 16*nt + l%16, half e = row K = 32*ks + 8*(l/16) + e (+8 if e>=8)
    v16h w1f[2][4], w2f[2][4];
    #pragma unroll
    for (int ks = 0; ks < 2; ++ks) {
        #pragma unroll
        for (int nt = 0; nt < 4; ++nt) {
            const float* s1 = W1 + (32 * ks) * 64 + 16 * nt + n0;
            const float* s2 = W2 + (32 * ks) * 64 + 16 * nt + n0;
            v16h t1, t2;
            #pragma unroll
            for (int e = 0; e < 16; ++e) {
                const int kk = 8 * hi + e + ((e >= 8) ? 8 : 0);
                t1[e] = (_Float16)s1[kk * 64];
                t2[e] = (_Float16)s2[kk * 64];
            }
            w1f[ks][nt] = t1;
            w2f[ks][nt] = t2;
        }
    }
    // per-lane bias / W3 values for column N = 16*nt + n0
    float b1v[4], b2v[4], w3v[4];
    #pragma unroll
    for (int nt = 0; nt < 4; ++nt) {
        int col = 16 * nt + n0;
        b1v[nt] = b1[col];
        b2v[nt] = b2[col];
        w3v[nt] = W3[col];
    }
    const float b3v = b3[0];

    // LDS: per-wave layer-1 activation tile (row-padded to 80 halfs = 160B, 16B aligned chunks)
    __shared__ _Float16 h1sh[WAVES_PER_BLOCK][16][80];
    __shared__ float    rsh[WAVES_PER_BLOCK][16];

    const int jt       = N >> 4;                 // j-tiles per i (128)
    const int nTiles   = N * jt;                 // 262144
    const int gwave    = blockIdx.x * WAVES_PER_BLOCK + wave;
    const int strideW  = gridDim.x * WAVES_PER_BLOCK;

    float acc = 0.0f;

    for (int t = gwave; t < nTiles; t += strideW) {
        const int i  = t >> 7;                   // t / jt   (jt == 128)
        const int j0 = (t & (jt - 1)) << 4;
        if (j0 + 15 <= i) continue;              // whole tile below strict upper triangle

        __builtin_amdgcn_wave_barrier();

        // ---- distance for this lane's pair (m = n0, j = j0+n0) ----
        const int j = j0 + n0;
        const float xi = R[3 * i],  yi = R[3 * i + 1],  zi = R[3 * i + 2];
        const float dx = xi - R[3 * j];
        const float dy = yi - R[3 * j + 1];
        const float dz = zi - R[3 * j + 2];
        const float r2 = dx * dx + dy * dy + dz * dz;
        const float r  = (r2 > 0.0f) ? sqrtf(r2) : 0.0f;
        if (hi == 0) rsh[wave][n0] = r;

        // ---- A fragment, kstep 0: [emb_i | emb_j] halves ----
        const int zi_ = Z[i];
        const int zj_ = Z[j];
        const float* ei = emb + zi_ * 16 + 8 * hi;
        const float* ej = emb + zj_ * 16 + 8 * hi;
        v16h a0;
        #pragma unroll
        for (int e = 0; e < 8; ++e) {
            a0[e]     = (_Float16)ei[e];   // K = 8*hi + e        (feat 0..15)
            a0[8 + e] = (_Float16)ej[e];   // K = 16 + 8*hi + e   (feat 16..31)
        }
        // ---- A fragment, kstep 1: RBF(r) ----
        v16h a1;
        #pragma unroll
        for (int e = 0; e < 16; ++e) {
            int rb = 8 * hi + e + ((e >= 8) ? 8 : 0);   // rbf index 0..31
            float d = r - (float)rb * CSPACE;
            a1[e] = (_Float16)__expf(-GAMMA_F * d * d);
        }

        // ---- Layer 1: bias folded into accumulator init (C cols = 16*nt+n0) ----
        v8f c1[4];
        #pragma unroll
        for (int nt = 0; nt < 4; ++nt) {
            const float b = b1v[nt];
            v8f c = {b, b, b, b, b, b, b, b};
            c = __builtin_amdgcn_wmma_f32_16x16x32_f16(false, a0, false, w1f[0][nt],
                                                       (short)0, c, false, false);
            c = __builtin_amdgcn_wmma_f32_16x16x32_f16(false, a1, false, w1f[1][nt],
                                                       (short)0, c, false, false);
            c1[nt] = c;
        }

        // ---- tanh -> LDS (C layout: reg rr, lane -> M=rr+8*hi, Ncol=16*nt+n0) ----
        #pragma unroll
        for (int nt = 0; nt < 4; ++nt) {
            #pragma unroll
            for (int rr = 0; rr < 8; ++rr) {
                float h = fast_tanh(c1[nt][rr]);
                h1sh[wave][rr + 8 * hi][16 * nt + n0] = (_Float16)h;
            }
        }
        __builtin_amdgcn_wave_barrier();   // keep LDS stores before re-load (in-order per wave)

        // ---- rebuild A fragments for layer 2 from LDS (two 16B ds loads per kstep) ----
        v16h a2[2];
        #pragma unroll
        for (int ks = 0; ks < 2; ++ks) {
            const v8h lo = *(const v8h*)&h1sh[wave][n0][32 * ks + 8 * hi];
            const v8h hc = *(const v8h*)&h1sh[wave][n0][32 * ks + 8 * hi + 16];
            v16h a;
            #pragma unroll
            for (int e = 0; e < 8; ++e) { a[e] = lo[e]; a[8 + e] = hc[e]; }
            a2[ks] = a;
        }

        // ---- Layer 2: bias folded into accumulator init ----
        v8f c2[4];
        #pragma unroll
        for (int nt = 0; nt < 4; ++nt) {
            const float b = b2v[nt];
            v8f c = {b, b, b, b, b, b, b, b};
            c = __builtin_amdgcn_wmma_f32_16x16x32_f16(false, a2[0], false, w2f[0][nt],
                                                       (short)0, c, false, false);
            c = __builtin_amdgcn_wmma_f32_16x16x32_f16(false, a2[1], false, w2f[1][nt],
                                                       (short)0, c, false, false);
            c2[nt] = c;
        }

        // ---- Layer 3 + mask: pair_E[m] = sum_N tanh(c2)[m][N]*W3[N] + b3 ----
        #pragma unroll
        for (int rr = 0; rr < 8; ++rr) {
            const int mm = rr + 8 * hi;
            const float rm = rsh[wave][mm];
            if ((i < j0 + mm) && (rm < CUTOFF_F)) {
                float s = (n0 == 0) ? b3v : 0.0f;   // add b3 exactly once per pair
                #pragma unroll
                for (int nt = 0; nt < 4; ++nt)
                    s += fast_tanh(c2[nt][rr]) * w3v[nt];
                acc += s;
            }
        }
    }

    // ---- full-wave reduction (covers both N halves and all M rows) ----
    #pragma unroll
    for (int off = 16; off > 0; off >>= 1)
        acc += __shfl_xor(acc, off, 32);
    if (lane == 0)
        atomicAdd(out, acc);
}

extern "C" void kernel_launch(void* const* d_in, const int* in_sizes, int n_in,
                              void* d_out, int out_size, void* d_ws, size_t ws_size,
                              hipStream_t stream) {
    const float* R   = (const float*)d_in[0];
    const int*   Z   = (const int*)  d_in[1];
    const float* emb = (const float*)d_in[2];
    const float* W1  = (const float*)d_in[3];
    const float* b1  = (const float*)d_in[4];
    const float* W2  = (const float*)d_in[5];
    const float* b2  = (const float*)d_in[6];
    const float* W3  = (const float*)d_in[7];
    const float* b3  = (const float*)d_in[8];
    float* out = (float*)d_out;

    const int N = in_sizes[0] / 3;   // 2048

    zero_kernel<<<1, 64, 0, stream>>>(out);
    pair_mlp_wmma<<<2048, WAVES_PER_BLOCK * 32, 0, stream>>>(
        R, Z, emb, W1, b1, W2, b2, W3, b3, out, N);
}
